// UMC_Loss_51342039056870
// MI455X (gfx1250) — matrix operand order repaired and checked by reading, not compiled, not run on hardware
//
#include <hip/hip_runtime.h>
#include <hip/hip_bf16.h>

typedef __bf16 bf16;
typedef __attribute__((ext_vector_type(16))) __bf16 v16bf;
typedef __attribute__((ext_vector_type(8)))  __bf16 v8bf;
typedef __attribute__((ext_vector_type(8)))  float   v8f;
typedef __attribute__((ext_vector_type(4)))  int     v4i;

constexpr int   BB      = 4;
constexpr int   NN      = 2048;
constexpr int   DD      = 64;
constexpr float EPSC    = 0.1f;
constexpr float TAUC    = 0.001f;
constexpr float FI      = TAUC / (TAUC + EPSC);     // damping exponent
constexpr float MU      = 1.0f / (float)NN;
constexpr float INV_EPS = 1.0f / EPSC;
constexpr size_t KELEM  = (size_t)BB * NN * NN;     // 16,777,216

// ---- gfx1250 async global->LDS copy (ASYNCcnt-tracked), with fallback ----
#if defined(__has_builtin)
#if __has_builtin(__builtin_amdgcn_global_load_async_to_lds_b128) && \
    __has_builtin(__builtin_amdgcn_s_wait_asynccnt)
#define USE_ASYNC 1
#endif
#endif
#ifndef USE_ASYNC
#define USE_ASYNC 0
#endif

__device__ __forceinline__ void cp16_async(bf16* dst_lds, const bf16* src_g) {
#if USE_ASYNC
    __builtin_amdgcn_global_load_async_to_lds_b128(
        (__attribute__((address_space(1))) v4i*)(src_g),
        (__attribute__((address_space(3))) v4i*)(dst_lds), 0, 0);
#else
    *(v8bf*)dst_lds = *(const v8bf*)src_g;
#endif
}

#if USE_ASYNC
#define ASYNC_WAIT(n) __builtin_amdgcn_s_wait_asynccnt(n)
#else
#define ASYNC_WAIT(n) ((void)0)
#endif

// ---------------------------------------------------------------- init
__global__ void k_init(float* __restrict__ out, float* __restrict__ uv, int count) {
    int i = blockIdx.x * blockDim.x + threadIdx.x;
    if (i == 0) out[0] = 0.0f;
    if (i < count) uv[i] = 1.0f;
}

// ------------------------------------------------- row-normalize -> bf16
__global__ __launch_bounds__(256) void k_rownorm(const float* __restrict__ x,
                                                 bf16* __restrict__ xn) {
    int wave = threadIdx.x >> 5, lane = threadIdx.x & 31;
    int row  = blockIdx.x * 8 + wave;
    const float* p = x + (size_t)row * DD;
    float a = p[lane], b = p[lane + 32];
    float s = a * a + b * b;
#pragma unroll
    for (int off = 16; off; off >>= 1) s += __shfl_xor(s, off, 32);
    float sc = 1.0f / (sqrtf(s) + 1e-8f);
    bf16* q = xn + (size_t)row * DD;
    q[lane]      = (bf16)(a * sc);
    q[lane + 32] = (bf16)(b * sc);
}

// ------------------------------------- K = exp((Xn·Ynᵀ − 1)/eps), bf16 out
// 128x128 tile per block (256 thr = 8 waves), K-dim = 64; async-staged LDS
__global__ __launch_bounds__(256) void k_costK(const bf16* __restrict__ X,
                                               const bf16* __restrict__ Y,
                                               bf16* __restrict__ Kout) {
    constexpr int LD = 72;                       // 64 + 8 pad halves
    __shared__ bf16 As[128 * LD];
    __shared__ bf16 Bs[128 * LD];
    int b  = blockIdx.z;
    int i0 = blockIdx.y * 128, j0 = blockIdx.x * 128;
    int t  = threadIdx.x;
    const bf16* Xb = X + (size_t)b * NN * DD;
    const bf16* Yb = Y + (size_t)b * NN * DD;
    {   // stage 128x64 of each operand; 2 threads/row, 32 halves each
        int r = t >> 1, c = (t & 1) * 32;
        const bf16* gx = Xb + (size_t)(i0 + r) * DD + c;
        bf16*       lx = As + r * LD + c;
        cp16_async(lx,      gx);
        cp16_async(lx + 8,  gx + 8);
        cp16_async(lx + 16, gx + 16);
        cp16_async(lx + 24, gx + 24);
        const bf16* gy = Yb + (size_t)(j0 + r) * DD + c;
        bf16*       ly = Bs + r * LD + c;
        cp16_async(ly,      gy);
        cp16_async(ly + 8,  gy + 8);
        cp16_async(ly + 16, gy + 16);
        cp16_async(ly + 24, gy + 24);
    }
    ASYNC_WAIT(0);
    __syncthreads();

    int wave = t >> 5, lane = t & 31, lr = lane & 15, hi = lane >> 4;
    int m_off = (wave >> 1) * 32;                // 0,32,64,96
    int n_off = (wave & 1) * 64;                 // 0,64
    v8f acc[2][4] = {};
#pragma unroll
    for (int kk = 0; kk < 64; kk += 32) {
        v16bf afr[2], bfr[4];
#pragma unroll
        for (int mi = 0; mi < 2; ++mi) {
            const bf16* pa = As + (m_off + mi * 16 + lr) * LD + kk + hi * 8;
            v8bf lo = *(const v8bf*)pa;
            v8bf hh = *(const v8bf*)(pa + 16);
            afr[mi] = __builtin_shufflevector(lo, hh, 0,1,2,3,4,5,6,7,8,9,10,11,12,13,14,15);
        }
#pragma unroll
        for (int ni = 0; ni < 4; ++ni) {
            const bf16* pb = Bs + (n_off + ni * 16 + lr) * LD + kk + hi * 8;
            v8bf lo = *(const v8bf*)pb;
            v8bf hh = *(const v8bf*)(pb + 16);
            bfr[ni] = __builtin_shufflevector(lo, hh, 0,1,2,3,4,5,6,7,8,9,10,11,12,13,14,15);
        }
#pragma unroll
        for (int mi = 0; mi < 2; ++mi)
#pragma unroll
            for (int ni = 0; ni < 4; ++ni)
                acc[mi][ni] = __builtin_amdgcn_wmma_f32_16x16x32_bf16(
                    false, afr[mi], false, bfr[ni], (short)0, acc[mi][ni], false, false);
    }
    bf16* Kb = Kout + (size_t)b * NN * NN;
#pragma unroll
    for (int mi = 0; mi < 2; ++mi)
#pragma unroll
        for (int ni = 0; ni < 4; ++ni)
#pragma unroll
            for (int r = 0; r < 8; ++r) {
                int gi = i0 + m_off + mi * 16 + hi * 8 + r;
                int gj = j0 + n_off + ni * 16 + lr;
                float s = acc[mi][ni][r];
                Kb[(size_t)gi * NN + gj] = (bf16)__expf((s - 1.0f) * INV_EPS);
            }
}

// ---------------------------- Sinkhorn: v = (nu / (Kᵀu + 1e-30))^fi
__global__ __launch_bounds__(256) void k_vupdate(const bf16* __restrict__ K,
                                                 const float* __restrict__ u,
                                                 float* __restrict__ v) {
    __shared__ float su[NN];
    int b = blockIdx.y, t = threadIdx.x;
    for (int i = t; i < NN; i += 256) su[i] = u[b * NN + i];
    __syncthreads();
    int m = blockIdx.x * 256 + t;
    const bf16* Kb = K + (size_t)b * NN * NN + m;
    float s = 0.0f;
#pragma unroll 8
    for (int n = 0; n < NN; ++n) {
        if ((n & 63) == 0) __builtin_prefetch(Kb + (size_t)(n + 64) * NN, 0, 0);
        s += (float)Kb[(size_t)n * NN] * su[n];
    }
    v[b * NN + m] = __powf(MU / (s + 1e-30f), FI);
}

// ---------------------------- Sinkhorn: u = (mu / (K v + 1e-30))^fi
__global__ __launch_bounds__(256) void k_uupdate(const bf16* __restrict__ K,
                                                 const float* __restrict__ v,
                                                 float* __restrict__ u) {
    __shared__ float sv[NN];
    int b = blockIdx.y, t = threadIdx.x;
    for (int i = t; i < NN; i += 256) sv[i] = v[b * NN + i];
    __syncthreads();
    int wave = t >> 5, lane = t & 31;
    int row  = blockIdx.x * 8 + wave;
    const bf16* Kb = K + (size_t)b * NN * NN + (size_t)row * NN;
    float s = 0.0f;
    for (int mb = lane * 8; mb < NN; mb += 256) {
        v8bf kv = *(const v8bf*)(Kb + mb);
#pragma unroll
        for (int j = 0; j < 8; ++j) s += (float)kv[j] * sv[mb + j];
    }
#pragma unroll
    for (int off = 16; off; off >>= 1) s += __shfl_xor(s, off, 32);
    if (lane == 0) u[b * NN + row] = __powf(MU / (s + 1e-30f), FI);
}

// ------------------- fold w = v1∘u2 into K1 columns (in place): A = K1·diag(w)
__global__ __launch_bounds__(256) void k_scaleA(bf16* __restrict__ K1,
                                                const float* __restrict__ v1,
                                                const float* __restrict__ u2) {
    size_t i8 = ((size_t)blockIdx.x * 256 + threadIdx.x) * 8;
    int b  = (int)(i8 >> 22);
    int m0 = (int)(i8 & (NN - 1));
    v8bf kv = *(const v8bf*)(K1 + i8);
#pragma unroll
    for (int j = 0; j < 8; ++j) {
        float w = v1[b * NN + m0 + j] * u2[b * NN + m0 + j];
        kv[j] = (bf16)((float)kv[j] * w);
    }
    *(v8bf*)(K1 + i8) = kv;
}

// ---------------------------------------------- K2t[b][j][k] = K2[b][k][j]
__global__ __launch_bounds__(256) void k_transpose(const bf16* __restrict__ src,
                                                   bf16* __restrict__ dst) {
    __shared__ bf16 tile[32][33];
    int b  = blockIdx.z;
    int tx = threadIdx.x & 31, ty = threadIdx.x >> 5;     // ty 0..7
    const bf16* S = src + (size_t)b * NN * NN;
    bf16*       T = dst + (size_t)b * NN * NN;
    int x = blockIdx.x * 32 + tx;
#pragma unroll
    for (int i = 0; i < 4; ++i)
        tile[ty + 8 * i][tx] = S[(size_t)(blockIdx.y * 32 + ty + 8 * i) * NN + x];
    __syncthreads();
    int xo = blockIdx.y * 32 + tx;
#pragma unroll
    for (int i = 0; i < 4; ++i)
        T[(size_t)(blockIdx.x * 32 + ty + 8 * i) * NN + xo] = tile[tx][ty + 8 * i];
}

// ---- f_indirect tile = u1⊗(A·K2)⊗v2 ; loss += mean|u3⊗K3⊗v3 − f_indirect|
// double-buffered async-to-LDS pipelined GEMM
__global__ __launch_bounds__(256) void k_final(const bf16* __restrict__ A,    // K1·diag(w)
                                               const bf16* __restrict__ Bt,   // K2 transposed
                                               const bf16* __restrict__ K3,
                                               const float* __restrict__ u1,
                                               const float* __restrict__ v2,
                                               const float* __restrict__ u3,
                                               const float* __restrict__ v3,
                                               float* __restrict__ out) {
    constexpr int LD = 40;                       // 32 + 8 pad halves
    __shared__ bf16  As[2][128 * LD];
    __shared__ bf16  Bs[2][128 * LD];
    __shared__ float s_u1[128], s_v2[128], s_u3[128], s_v3[128];
    __shared__ float red[256];
    int b  = blockIdx.z;
    int i0 = blockIdx.y * 128, j0 = blockIdx.x * 128;
    int t  = threadIdx.x;
    if (t < 128) {
        s_u1[t] = u1[b * NN + i0 + t];
        s_u3[t] = u3[b * NN + i0 + t];
    } else {
        int tt = t - 128;
        s_v2[tt] = v2[b * NN + j0 + tt];
        s_v3[tt] = v3[b * NN + j0 + tt];
    }
    const bf16* Ab = A  + (size_t)b * NN * NN;
    const bf16* Bb = Bt + (size_t)b * NN * NN;
    int wave = t >> 5, lane = t & 31, lr = lane & 15, hi = lane >> 4;
    int m_off = (wave >> 1) * 32, n_off = (wave & 1) * 64;
    int sr = t >> 1, sc = (t & 1) * 16;
    const bf16* gA = Ab + (size_t)(i0 + sr) * NN + sc;   // + kk at issue time
    const bf16* gB = Bb + (size_t)(j0 + sr) * NN + sc;
    int lofs = sr * LD + sc;
    v8f acc[2][4] = {};

    auto issue = [&](int kk, int buf) {
        cp16_async(&As[buf][lofs],     gA + kk);
        cp16_async(&As[buf][lofs + 8], gA + kk + 8);
        cp16_async(&Bs[buf][lofs],     gB + kk);
        cp16_async(&Bs[buf][lofs + 8], gB + kk + 8);
    };

    issue(0, 0);                                  // prologue
    for (int i = 0; i < NN / 32; ++i) {
        int cur = i & 1;
        if (i + 1 < NN / 32) {
            issue(32 * (i + 1), cur ^ 1);         // prefetch next tile into other buffer
            ASYNC_WAIT(4);                        // 4 newest in flight; current tile done
        } else {
            ASYNC_WAIT(0);
        }
        __syncthreads();

        v16bf afr[2], bfr[4];
#pragma unroll
        for (int mi = 0; mi < 2; ++mi) {
            const bf16* pa = &As[cur][(m_off + mi * 16 + lr) * LD + hi * 8];
            v8bf lo = *(const v8bf*)pa;
            v8bf hh = *(const v8bf*)(pa + 16);
            afr[mi] = __builtin_shufflevector(lo, hh, 0,1,2,3,4,5,6,7,8,9,10,11,12,13,14,15);
        }
#pragma unroll
        for (int ni = 0; ni < 4; ++ni) {
            const bf16* pb = &Bs[cur][(n_off + ni * 16 + lr) * LD + hi * 8];
            v8bf lo = *(const v8bf*)pb;
            v8bf hh = *(const v8bf*)(pb + 16);
            bfr[ni] = __builtin_shufflevector(lo, hh, 0,1,2,3,4,5,6,7,8,9,10,11,12,13,14,15);
        }
#pragma unroll
        for (int mi = 0; mi < 2; ++mi)
#pragma unroll
            for (int ni = 0; ni < 4; ++ni)
                acc[mi][ni] = __builtin_amdgcn_wmma_f32_16x16x32_bf16(
                    false, afr[mi], false, bfr[ni], (short)0, acc[mi][ni], false, false);
        __syncthreads();                          // readers done before buffer reuse
    }

    const bf16* K3b = K3 + (size_t)b * NN * NN;
    float loc = 0.0f;
#pragma unroll
    for (int mi = 0; mi < 2; ++mi)
#pragma unroll
        for (int ni = 0; ni < 4; ++ni)
#pragma unroll
            for (int r = 0; r < 8; ++r) {
                int li = m_off + mi * 16 + hi * 8 + r;
                int lj = n_off + ni * 16 + lr;
                float ind = s_u1[li] * s_v2[lj] * acc[mi][ni][r];
                float dir = s_u3[li] * (float)K3b[(size_t)(i0 + li) * NN + (j0 + lj)] * s_v3[lj];
                loc += fabsf(dir - ind);
            }
    red[t] = loc;
    __syncthreads();
    for (int off = 128; off; off >>= 1) {
        if (t < off) red[t] += red[t + off];
        __syncthreads();
    }
    if (t == 0) atomicAdd(out, red[0] * (1.0f / (float)KELEM));
}

// ----------------------------------------------------------------- driver
extern "C" void kernel_launch(void* const* d_in, const int* in_sizes, int n_in,
                              void* d_out, int out_size, void* d_ws, size_t ws_size,
                              hipStream_t stream) {
    (void)in_sizes; (void)n_in; (void)out_size; (void)ws_size;
    const float* src = (const float*)d_in[0];
    const float* tgt = (const float*)d_in[1];
    const float* gen = (const float*)d_in[2];
    float* out = (float*)d_out;

    char* ws = (char*)d_ws;
    const size_t KB = KELEM * sizeof(bf16);          // 33.55 MB per K matrix
    bf16* K1  = (bf16*)(ws);
    bf16* K2  = (bf16*)(ws + KB);
    bf16* K3  = (bf16*)(ws + 2 * KB);
    bf16* K2t = (bf16*)(ws + 3 * KB);
    bf16* xnS = (bf16*)(ws + 4 * KB);
    bf16* xnT = xnS + (size_t)BB * NN * DD;
    bf16* xnG = xnT + (size_t)BB * NN * DD;
    float* uv = (float*)(xnG + (size_t)BB * NN * DD);
    float* u1 = uv, *v1 = uv + BB * NN, *u2 = uv + 2 * BB * NN,
         *v2 = uv + 3 * BB * NN, *u3 = uv + 4 * BB * NN, *v3 = uv + 5 * BB * NN;

    k_init<<<(6 * BB * NN + 255) / 256, 256, 0, stream>>>(out, uv, 6 * BB * NN);
    k_rownorm<<<1024, 256, 0, stream>>>(src, xnS);
    k_rownorm<<<1024, 256, 0, stream>>>(tgt, xnT);
    k_rownorm<<<1024, 256, 0, stream>>>(gen, xnG);

    dim3 gtile(NN / 128, NN / 128, BB);              // 16x16x4
    k_costK<<<gtile, 256, 0, stream>>>(xnS, xnT, K1);
    k_costK<<<gtile, 256, 0, stream>>>(xnT, xnG, K2);
    k_costK<<<gtile, 256, 0, stream>>>(xnS, xnG, K3);

    dim3 gv(NN / 256, BB), gu(NN / 8, BB);
    for (int it = 0; it < 50; ++it) {
        k_vupdate<<<gv, 256, 0, stream>>>(K1, u1, v1);
        k_uupdate<<<gu, 256, 0, stream>>>(K1, v1, u1);
    }
    for (int it = 0; it < 50; ++it) {
        k_vupdate<<<gv, 256, 0, stream>>>(K2, u2, v2);
        k_uupdate<<<gu, 256, 0, stream>>>(K2, v2, u2);
    }
    for (int it = 0; it < 50; ++it) {
        k_vupdate<<<gv, 256, 0, stream>>>(K3, u3, v3);
        k_uupdate<<<gu, 256, 0, stream>>>(K3, v3, u3);
    }

    k_scaleA<<<(int)(KELEM / 8 / 256), 256, 0, stream>>>(K1, v1, u2);
    k_transpose<<<dim3(NN / 32, NN / 32, BB), 256, 0, stream>>>(K2, K2t);
    k_final<<<gtile, 256, 0, stream>>>(K1, K2t, K3, u1, v2, u3, v3, out);
}
